// LanguageModel_88390426952040
// MI455X (gfx1250) — matrix operand order repaired and checked by reading, not compiled, not run on hardware
//
#include <hip/hip_runtime.h>
#include <hip/hip_bf16.h>
#include <math.h>

typedef __attribute__((ext_vector_type(16))) _Float16 v16h;
typedef __attribute__((ext_vector_type(8)))  _Float16 v8h;
typedef __attribute__((ext_vector_type(4)))  _Float16 v4h;
typedef __attribute__((ext_vector_type(8)))  float    v8f;

// ---------------- WMMA fragment helpers (wave32, 16x16x32 f16) ----------------
// A 16x32 (MxK): lane L -> row M = L%16, K = 8*(L/16)+e (e<8), +8 more for e>=8.
// B 32x16 (KxN): lane L -> col N = L%16, K = 8*(L/16)+e (e<8), +8 more for e>=8.  B is staged
// transposed in LDS as Bt[N][K] so both gathers are two contiguous 16B ds loads.

__device__ inline v16h cat16(v8h lo, v8h hi) {
    v16h r;
#pragma unroll
    for (int i = 0; i < 8; ++i) { r[i] = lo[i]; r[i + 8] = hi[i]; }
    return r;
}

__device__ inline v16h frag_a(const _Float16* base, int stride) {
    int lane = threadIdx.x & 31;
    int r = lane & 15, g = lane >> 4;
    const _Float16* p = base + r * stride + 8 * g;
    return cat16(*(const v8h*)p, *(const v8h*)(p + 16));
}

__device__ inline v16h frag_b(const _Float16* base, int stride) {
    int lane = threadIdx.x & 31;
    int n = lane & 15, g = lane >> 4;
    const _Float16* p = base + n * stride + 8 * g;
    return cat16(*(const v8h*)p, *(const v8h*)(p + 16));
}

__device__ inline v8f wmma32(v16h a, v16h b, v8f c) {
    return __builtin_amdgcn_wmma_f32_16x16x32_f16(false, a, false, b, (short)0, c,
                                                  false, false);
}

// ---------------- Generic WMMA GEMM: C = act(A*B + bias + res) ----------------
// A: f16 [M,K] row-major (lda). B: f16 [K,N] row-major (ldb).
// Block tile 128x64x32, 8 waves, each wave owns 32x32 (2x2 WMMA tiles).
__global__ __launch_bounds__(256) void gemm_f16(
    const _Float16* __restrict__ A, int lda,
    const _Float16* __restrict__ Bm, int ldb,
    const float* __restrict__ bias, const float* __restrict__ res,
    float* __restrict__ out32, int ld32,
    _Float16* __restrict__ out16, int ld16,
    int M, int N, int Nstore, int K, int relu)
{
    __shared__ alignas(16) _Float16 As[128 * 32];
    __shared__ alignas(16) _Float16 Bs[64 * 32];
    const int tid = threadIdx.x;
    const int m0 = blockIdx.y * 128, n0 = blockIdx.x * 64;
    const int w = tid >> 5, lane = tid & 31;
    const int wm = w >> 1, wn = w & 1;

    v8f acc[2][2] = {};
    for (int k0 = 0; k0 < K; k0 += 32) {
        // A tile: straight wide copy, 512 chunks of 8 halfs (16B)
        for (int c = tid; c < 512; c += 256) {
            int m = c >> 2, kk = (c & 3) * 8;
            *(v8h*)&As[m * 32 + kk] =
                *(const v8h*)&A[(size_t)(m0 + m) * lda + k0 + kk];
        }
        // B tile: wide loads along N, transpose into Bt[N][K] via LDS stores
        for (int c = tid; c < 256; c += 256) {
            int kk = c >> 3, n8 = (c & 7) * 8;
            int gn = n0 + n8;
            v8h v;
            if (gn + 7 < N) {
                v = *(const v8h*)&Bm[(size_t)(k0 + kk) * ldb + gn];
            } else {
#pragma unroll
                for (int i = 0; i < 8; ++i)
                    v[i] = (gn + i < N)
                               ? Bm[(size_t)(k0 + kk) * ldb + gn + i]
                               : (_Float16)0.f;
            }
#pragma unroll
            for (int i = 0; i < 8; ++i) Bs[(n8 + i) * 32 + kk] = v[i];
        }
        __syncthreads();
        v16h a0 = frag_a(As + (wm * 32) * 32, 32);
        v16h a1 = frag_a(As + (wm * 32 + 16) * 32, 32);
        v16h b0 = frag_b(Bs + (wn * 32) * 32, 32);
        v16h b1 = frag_b(Bs + (wn * 32 + 16) * 32, 32);
        acc[0][0] = wmma32(a0, b0, acc[0][0]);
        acc[0][1] = wmma32(a0, b1, acc[0][1]);
        acc[1][0] = wmma32(a1, b0, acc[1][0]);
        acc[1][1] = wmma32(a1, b1, acc[1][1]);
        __syncthreads();
    }
    const int nl = lane & 15, g = lane >> 4;
    for (int ti = 0; ti < 2; ++ti)
        for (int tj = 0; tj < 2; ++tj) {
            int n = n0 + wn * 32 + tj * 16 + nl;
            if (n >= Nstore) continue;
            float bv = bias ? bias[n] : 0.f;
#pragma unroll
            for (int j = 0; j < 8; ++j) {
                int m = m0 + wm * 32 + ti * 16 + j + 8 * g;
                float v = acc[ti][tj][j] + bv;
                if (res) v += res[(size_t)m * ld32 + n];
                if (relu) v = fmaxf(v, 0.f);
                if (out32) out32[(size_t)m * ld32 + n] = v;
                if (out16) out16[(size_t)m * ld16 + n] = (_Float16)v;
            }
        }
}

// ---------------- Fused causal attention for one (b,h,64-row slab) -----------
// qkv layout: [B*S, 1152] f16, cols: q=h*64, k=384+h*64, v=768+h*64.
__global__ __launch_bounds__(256) void attn_kernel(const _Float16* __restrict__ qkv,
                                                   _Float16* __restrict__ a16)
{
    __shared__ alignas(16) _Float16 Qs[64 * 64];
    __shared__ alignas(16) _Float16 Ks[64 * 64];
    __shared__ alignas(16) _Float16 Vs[64 * 64];   // stored transposed: [e][t]
    __shared__ alignas(16) _Float16 Ps[64 * 256];
    const int tid = threadIdx.x;
    const int rb = blockIdx.x, h = blockIdx.y, b = blockIdx.z;
    const int s0 = rb * 64;
    const _Float16* base = qkv + (size_t)b * 256 * 1152;
    const int qc = h * 64, kc = 384 + h * 64, vc = 768 + h * 64;

    // Q tile: 512 chunks of 8 halfs
    for (int c = tid; c < 512; c += 256) {
        int r = c >> 3, e = (c & 7) * 8;
        *(v8h*)&Qs[r * 64 + e] =
            *(const v8h*)&base[(size_t)(s0 + r) * 1152 + qc + e];
    }
    const int w = tid >> 5, lane = tid & 31;
    const int nl = lane & 15, g = lane >> 4;

    // Phase 1: S = (Q Kᵀ)*scale with causal mask -> Ps (f16)
    for (int t0 = 0; t0 < 256; t0 += 64) {
        __syncthreads();
        for (int c = tid; c < 512; c += 256) {
            int r = c >> 3, e = (c & 7) * 8;
            *(v8h*)&Ks[r * 64 + e] =
                *(const v8h*)&base[(size_t)(t0 + r) * 1152 + kc + e];
        }
        __syncthreads();
#pragma unroll
        for (int q2 = 0; q2 < 2; ++q2) {
            int tt = 2 * w + q2, tm = tt >> 2, tn = tt & 3;
            v8f acc = {};
            acc = wmma32(frag_a(Qs + (tm * 16) * 64, 64),
                         frag_b(Ks + (tn * 16) * 64, 64), acc);
            acc = wmma32(frag_a(Qs + (tm * 16) * 64 + 32, 64),
                         frag_b(Ks + (tn * 16) * 64 + 32, 64), acc);
            int tcol = t0 + tn * 16 + nl;
#pragma unroll
            for (int j = 0; j < 8; ++j) {
                int m = tm * 16 + j + 8 * g;
                float sv = acc[j] * 0.125f;  // 1/sqrt(64)
                Ps[m * 256 + tcol] =
                    (tcol <= s0 + m) ? (_Float16)sv : (_Float16)(-60000.f);
            }
        }
    }
    __syncthreads();
    // Row softmax (64 rows, f32 math)
    if (tid < 64) {
        int r = tid;
        float mx = -1e30f;
        for (int t = 0; t < 256; ++t) mx = fmaxf(mx, (float)Ps[r * 256 + t]);
        float sum = 0.f;
        for (int t = 0; t < 256; ++t) {
            float e = __expf((float)Ps[r * 256 + t] - mx);
            sum += e;
            Ps[r * 256 + t] = (_Float16)e;
        }
        float inv = 1.f / sum;
        for (int t = 0; t < 256; ++t)
            Ps[r * 256 + t] = (_Float16)((float)Ps[r * 256 + t] * inv);
    }
    // Phase 2: O = P V
    v8f acc2[2] = {};
    for (int t0 = 0; t0 < 256; t0 += 64) {
        __syncthreads();
        for (int c = tid; c < 512; c += 256) {
            int tl = c >> 3, e8 = (c & 7) * 8;
            v8h v = *(const v8h*)&base[(size_t)(t0 + tl) * 1152 + vc + e8];
#pragma unroll
            for (int i = 0; i < 8; ++i) Vs[(e8 + i) * 64 + tl] = v[i];
        }
        __syncthreads();
#pragma unroll
        for (int q2 = 0; q2 < 2; ++q2) {
            int tt = 2 * w + q2, tm = tt >> 2, tn = tt & 3;
#pragma unroll
            for (int kk = 0; kk < 64; kk += 32) {
                acc2[q2] = wmma32(frag_a(Ps + (tm * 16) * 256 + t0 + kk, 256),
                                  frag_b(Vs + (tn * 16) * 64 + kk, 64), acc2[q2]);
            }
        }
    }
#pragma unroll
    for (int q2 = 0; q2 < 2; ++q2) {
        int tt = 2 * w + q2, tm = tt >> 2, tn = tt & 3;
#pragma unroll
        for (int j = 0; j < 8; ++j) {
            int m = tm * 16 + j + 8 * g;
            int row = b * 256 + s0 + m;
            int col = h * 64 + tn * 16 + nl;
            a16[(size_t)row * 384 + col] = (_Float16)acc2[q2][j];
        }
    }
}

// ---------------- LayerNorm (one wave32 per row of 384, float4 path) ---------
__global__ __launch_bounds__(256) void ln_f16(const float* __restrict__ x,
                                              const float* __restrict__ gam,
                                              const float* __restrict__ bet,
                                              _Float16* __restrict__ out, int rows)
{
    int row = blockIdx.x * 8 + (threadIdx.x >> 5);
    int lane = threadIdx.x & 31;
    if (row >= rows) return;
    const float4* xr = (const float4*)(x + (size_t)row * 384);
    float4 c[3];
    float s1 = 0.f, s2 = 0.f;
#pragma unroll
    for (int i = 0; i < 3; ++i) {
        c[i] = xr[lane + 32 * i];
        s1 += c[i].x + c[i].y + c[i].z + c[i].w;
        s2 += c[i].x * c[i].x + c[i].y * c[i].y + c[i].z * c[i].z + c[i].w * c[i].w;
    }
    for (int o = 16; o; o >>= 1) { s1 += __shfl_xor(s1, o, 32); s2 += __shfl_xor(s2, o, 32); }
    float m = s1 * (1.f / 384.f);
    float var = s2 * (1.f / 384.f) - m * m;
    float r = rsqrtf(var + 1e-5f);
    const float4* g4 = (const float4*)gam;
    const float4* b4 = (const float4*)bet;
    v4h* orow = (v4h*)(out + (size_t)row * 384);
#pragma unroll
    for (int i = 0; i < 3; ++i) {
        int ci = lane + 32 * i;
        float4 gv = g4[ci], bv = b4[ci], v = c[i];
        v4h o;
        o[0] = (_Float16)((v.x - m) * r * gv.x + bv.x);
        o[1] = (_Float16)((v.y - m) * r * gv.y + bv.y);
        o[2] = (_Float16)((v.z - m) * r * gv.z + bv.z);
        o[3] = (_Float16)((v.w - m) * r * gv.w + bv.w);
        orow[ci] = o;
    }
}

// ---------------- Embedding, weight packing, loss ----------------------------
__global__ void embed_k(const int* __restrict__ idx, const float* __restrict__ tok,
                        const float* __restrict__ pos, float* __restrict__ x)
{
    size_t t = (size_t)blockIdx.x * 256 + threadIdx.x;   // float4 granularity
    if (t >= (size_t)32768 * 96) return;
    int bs = (int)(t / 96), c = (int)(t % 96);
    int s = bs & 255;
    float4 tv = ((const float4*)tok)[(size_t)idx[bs] * 96 + c];
    float4 pv = ((const float4*)pos)[(size_t)s * 96 + c];
    float4 o = {tv.x + pv.x, tv.y + pv.y, tv.z + pv.z, tv.w + pv.w};
    ((float4*)x)[t] = o;
}

__global__ void pack_qkv_k(const float* __restrict__ Wq, const float* __restrict__ Wk,
                           const float* __restrict__ Wv, _Float16* __restrict__ out)
{
    size_t t = (size_t)blockIdx.x * 256 + threadIdx.x;
    if (t >= (size_t)6 * 384 * 1152) return;
    int c = (int)(t % 1152);
    size_t ld = t / 1152;
    int d = (int)(ld % 384), l = (int)(ld / 384);
    int which = c / 384, h = (c % 384) / 64, e = c % 64;
    const float* W = (which == 0) ? Wq : (which == 1) ? Wk : Wv;
    out[t] = (_Float16)W[(((size_t)l * 6 + h) * 384 + d) * 64 + e];
}

__global__ void f32_to_f16_k(const float* __restrict__ in, _Float16* __restrict__ out,
                             size_t n)
{
    size_t t = (size_t)blockIdx.x * 256 + threadIdx.x;
    if (t < n) out[t] = (_Float16)in[t];
}

__global__ void pack_wout_k(const float* __restrict__ W, _Float16* __restrict__ out)
{
    int t = blockIdx.x * 256 + threadIdx.x;
    if (t >= 384 * 80) return;
    int d = t / 80, c = t % 80;
    out[t] = (c < 65) ? (_Float16)W[d * 65 + c] : (_Float16)0.f;
}

__global__ void zero_k(float* p) { if (threadIdx.x == 0 && blockIdx.x == 0) p[0] = 0.f; }

__global__ __launch_bounds__(256) void loss_k(const float* __restrict__ logits,
                                              const int* __restrict__ tgt,
                                              float* __restrict__ loss)
{
    int row = blockIdx.x * 8 + (threadIdx.x >> 5);
    int lane = threadIdx.x & 31;
    const float* lr = logits + (size_t)row * 65;
    int t = tgt[row];
    float mx = -1e30f, tv = -1e30f;
    for (int j = lane; j < 65; j += 32) {
        float v = lr[j];
        mx = fmaxf(mx, v);
        if (j == t) tv = v;
    }
    for (int o = 16; o; o >>= 1) {
        mx = fmaxf(mx, __shfl_xor(mx, o, 32));
        tv = fmaxf(tv, __shfl_xor(tv, o, 32));
    }
    float sum = 0.f;
    for (int j = lane; j < 65; j += 32) sum += __expf(lr[j] - mx);
    for (int o = 16; o; o >>= 1) sum += __shfl_xor(sum, o, 32);
    if (lane == 0) atomicAdd(loss, (mx + logf(sum) - tv) * (1.0f / 32768.f));
}

// ---------------- Host orchestration -----------------------------------------
extern "C" void kernel_launch(void* const* d_in, const int* in_sizes, int n_in,
                              void* d_out, int out_size, void* d_ws, size_t ws_size,
                              hipStream_t stream)
{
    (void)in_sizes; (void)n_in; (void)out_size; (void)ws_size;
    const int*   idx     = (const int*)d_in[0];
    const int*   targets = (const int*)d_in[1];
    const float* tok  = (const float*)d_in[2];
    const float* pos  = (const float*)d_in[3];
    const float* Wq   = (const float*)d_in[4];
    const float* Wk   = (const float*)d_in[5];
    const float* Wv   = (const float*)d_in[6];
    const float* Wo   = (const float*)d_in[7];
    const float* bo   = (const float*)d_in[8];
    const float* ln1g = (const float*)d_in[9];
    const float* ln1b = (const float*)d_in[10];
    const float* ln2g = (const float*)d_in[11];
    const float* ln2b = (const float*)d_in[12];
    const float* W1   = (const float*)d_in[13];
    const float* b1   = (const float*)d_in[14];
    const float* W2   = (const float*)d_in[15];
    const float* b2   = (const float*)d_in[16];
    const float* lnfg = (const float*)d_in[17];
    const float* lnfb = (const float*)d_in[18];
    const float* Wout = (const float*)d_in[19];
    const float* bout = (const float*)d_in[20];

    float* outp   = (float*)d_out;
    float* logits = outp;
    float* loss   = outp + (size_t)32768 * 65;

    char* p = (char*)d_ws;
    auto alloc = [&](size_t bytes) -> void* {
        void* r = (void*)p;
        p += (bytes + 255) & ~(size_t)255;
        return r;
    };
    float*    x      = (float*)   alloc((size_t)32768 * 384 * 4);
    _Float16* h16    = (_Float16*)alloc((size_t)32768 * 384 * 2);
    _Float16* qkv16  = (_Float16*)alloc((size_t)32768 * 1152 * 2);
    _Float16* a16    = (_Float16*)alloc((size_t)32768 * 384 * 2);
    _Float16* mid16  = (_Float16*)alloc((size_t)32768 * 1536 * 2);
    _Float16* Wqkv16 = (_Float16*)alloc((size_t)6 * 384 * 1152 * 2);
    _Float16* Wo16   = (_Float16*)alloc((size_t)6 * 384 * 384 * 2);
    _Float16* W116   = (_Float16*)alloc((size_t)6 * 384 * 1536 * 2);
    _Float16* W216   = (_Float16*)alloc((size_t)6 * 1536 * 384 * 2);
    _Float16* Wout16 = (_Float16*)alloc((size_t)384 * 80 * 2);

    { size_t n = (size_t)6 * 384 * 1152;
      pack_qkv_k<<<(unsigned)((n + 255) / 256), 256, 0, stream>>>(Wq, Wk, Wv, Wqkv16); }
    { size_t n = (size_t)6 * 384 * 384;
      f32_to_f16_k<<<(unsigned)((n + 255) / 256), 256, 0, stream>>>(Wo, Wo16, n); }
    { size_t n = (size_t)6 * 384 * 1536;
      f32_to_f16_k<<<(unsigned)((n + 255) / 256), 256, 0, stream>>>(W1, W116, n); }
    { size_t n = (size_t)6 * 1536 * 384;
      f32_to_f16_k<<<(unsigned)((n + 255) / 256), 256, 0, stream>>>(W2, W216, n); }
    pack_wout_k<<<(384 * 80 + 255) / 256, 256, 0, stream>>>(Wout, Wout16);
    { size_t n = (size_t)32768 * 96;
      embed_k<<<(unsigned)((n + 255) / 256), 256, 0, stream>>>(idx, tok, pos, x); }

    for (int l = 0; l < 6; ++l) {
        ln_f16<<<4096, 256, 0, stream>>>(x, ln1g + l * 384, ln1b + l * 384, h16, 32768);

        dim3 gq((1152 + 63) / 64, 32768 / 128);
        gemm_f16<<<gq, 256, 0, stream>>>(h16, 384, Wqkv16 + (size_t)l * 384 * 1152,
                                         1152, nullptr, nullptr, nullptr, 0,
                                         qkv16, 1152, 32768, 1152, 1152, 384, 0);

        dim3 ga(4, 6, 128);
        attn_kernel<<<ga, 256, 0, stream>>>(qkv16, a16);

        dim3 go((384 + 63) / 64, 32768 / 128);
        gemm_f16<<<go, 256, 0, stream>>>(a16, 384, Wo16 + (size_t)l * 384 * 384,
                                         384, bo + l * 384, x, x, 384,
                                         nullptr, 0, 32768, 384, 384, 384, 0);

        ln_f16<<<4096, 256, 0, stream>>>(x, ln2g + l * 384, ln2b + l * 384, h16, 32768);

        dim3 g1((1536 + 63) / 64, 32768 / 128);
        gemm_f16<<<g1, 256, 0, stream>>>(h16, 384, W116 + (size_t)l * 384 * 1536,
                                         1536, b1 + l * 1536, nullptr, nullptr, 0,
                                         mid16, 1536, 32768, 1536, 1536, 384, 1);

        dim3 g2((384 + 63) / 64, 32768 / 128);
        gemm_f16<<<g2, 256, 0, stream>>>(mid16, 1536, W216 + (size_t)l * 1536 * 384,
                                         384, b2 + l * 384, x, x, 384,
                                         nullptr, 0, 32768, 384, 384, 1536, 0);
    }

    ln_f16<<<4096, 256, 0, stream>>>(x, lnfg, lnfb, h16, 32768);
    dim3 gl((80 + 63) / 64, 32768 / 128);
    gemm_f16<<<gl, 256, 0, stream>>>(h16, 384, Wout16, 80, bout, nullptr,
                                     logits, 65, nullptr, 0, 32768, 80, 65, 384, 0);

    zero_k<<<1, 64, 0, stream>>>(loss);
    loss_k<<<4096, 256, 0, stream>>>(logits, targets, loss);
}
